// EPSparseMoE_69037304316414
// MI455X (gfx1250) — matrix operand order repaired and checked by reading, not compiled, not run on hardware
//
#include <hip/hip_runtime.h>
#include <hip/hip_bf16.h>

// Problem constants (B=4, L=2048, D=1024, F=2048, E=8, TOP_K=2)
#define D_  1024
#define F_  2048
#define E_  8
#define N_  8192        // B*L tokens
#define MT  16          // token-tile rows (one WMMA M)
#define FCHUNK 256      // F processed per chunk (8 waves x 32 cols)

typedef __attribute__((ext_vector_type(16))) __bf16 v16bf;
typedef __attribute__((ext_vector_type(8)))  __bf16 v8bf;
typedef __attribute__((ext_vector_type(8)))  float  v8f;

// ---- CDNA5 WMMA bf16 operand builders -------------------------------------
// A matrix (16x32, MxK), 16-bit: lane<16 -> M=lane, j0..15 = K k0+0..15 ;
// lane>=16 -> M=lane-16, j0..15 = K k0+16..31 (contiguous, matches B).
__device__ __forceinline__ v16bf ld_a16(const __bf16* row, int k0, int lane) {
  const __bf16* p = row + k0 + ((lane & 16) ? 16 : 0);
  v8bf lo = *(const v8bf*)(p);
  v8bf hi = *(const v8bf*)(p + 8);
  v16bf r;
#pragma unroll
  for (int i = 0; i < 8; ++i) { r[i] = lo[i]; r[i + 8] = hi[i]; }
  return r;
}

// B matrix (32x16, KxN): lane holds column N=lane&15, K = j + (lane<16?0:16),
// j contiguous. From a PRE-TRANSPOSED bf16 weight (K contiguous per column):
// just two b128 loads per lane.
__device__ __forceinline__ v16bf ld_b_t(const __bf16* base, int ldr, int lane) {
  const __bf16* p = base + (size_t)(lane & 15) * ldr + ((lane & 16) ? 16 : 0);
  v8bf lo = *(const v8bf*)(p);
  v8bf hi = *(const v8bf*)(p + 8);
  v16bf r;
#pragma unroll
  for (int i = 0; i < 8; ++i) { r[i] = lo[i]; r[i + 8] = hi[i]; }
  return r;
}

// Fallback: gather a column from row-major fp32 W (row stride ld), cvt to bf16.
__device__ __forceinline__ v16bf ld_b_g32(const float* base, int ld, int lane) {
  const float* p = base + (size_t)((lane & 16) ? 16 : 0) * ld + (lane & 15);
  v16bf r;
#pragma unroll
  for (int j = 0; j < 16; ++j) r[j] = (__bf16)p[(size_t)j * ld];
  return r;
}

__device__ __forceinline__ float silu_fast(float g) {
  return g * __builtin_amdgcn_rcpf(1.f + __expf(-g));  // v_rcp_f32, no IEEE div
}

// ---- Kernel 0: zero the per-expert counters --------------------------------
__global__ void moe_init(int* cnt) {
  if (threadIdx.x < E_) cnt[threadIdx.x] = 0;
}

// ---- Pre-pass: fp32 [rows][cols] -> bf16 transposed [cols][rows] -----------
__global__ __launch_bounds__(256)
void transpose_cvt(const float* __restrict__ src, __bf16* __restrict__ dst,
                   int rows, int cols) {
  __shared__ float tile[32][33];
  src += (size_t)blockIdx.z * rows * cols;
  dst += (size_t)blockIdx.z * rows * cols;
  const int c0 = blockIdx.x * 32, r0 = blockIdx.y * 32;
  const int tx = threadIdx.x & 31, ty = threadIdx.x >> 5;   // 32x8
#pragma unroll
  for (int i = 0; i < 32; i += 8)
    tile[ty + i][tx] = src[(size_t)(r0 + ty + i) * cols + c0 + tx];
  __syncthreads();
#pragma unroll
  for (int i = 0; i < 32; i += 8)
    dst[(size_t)(c0 + ty + i) * rows + r0 + tx] = (__bf16)tile[tx][ty + i];
}

// ---- Kernel 1: router (logits, top-2 softmax, per-expert token lists) ------
__global__ void moe_router(const float* __restrict__ x, const float* __restrict__ gw,
                           float* __restrict__ logits, int* __restrict__ cnt,
                           int* __restrict__ idxb, float* __restrict__ wgtb) {
  const int wave = threadIdx.x >> 5;
  const int lane = threadIdx.x & 31;
  const int n = blockIdx.x * 8 + wave;
  const float* xr = x + (size_t)n * D_;

  float acc[E_];
#pragma unroll
  for (int e = 0; e < E_; ++e) acc[e] = 0.f;
  for (int d = lane; d < D_; d += 32) {
    float xv = xr[d];
    const float* g = gw + d * E_;
#pragma unroll
    for (int e = 0; e < E_; ++e) acc[e] += xv * g[e];
  }
#pragma unroll
  for (int e = 0; e < E_; ++e) {
    for (int off = 16; off > 0; off >>= 1) acc[e] += __shfl_xor(acc[e], off, 32);
  }
  if (lane < E_) logits[(size_t)n * E_ + lane] = acc[lane];

  if (lane == 0) {
    int i0 = 0; float m0 = acc[0];
#pragma unroll
    for (int e = 1; e < E_; ++e) if (acc[e] > m0) { m0 = acc[e]; i0 = e; }
    int i1 = (i0 == 0) ? 1 : 0; float m1 = acc[i1];
#pragma unroll
    for (int e = 0; e < E_; ++e) if (e != i0 && acc[e] > m1) { m1 = acc[e]; i1 = e; }
    float p0 = 1.f / (1.f + __expf(m1 - m0));   // softmax over selected pair
    float p1 = 1.f - p0;
    int s0 = atomicAdd(&cnt[i0], 1);
    idxb[i0 * N_ + s0] = n; wgtb[i0 * N_ + s0] = p0;
    int s1 = atomicAdd(&cnt[i1], 1);
    idxb[i1 * N_ + s1] = n; wgtb[i1 * N_ + s1] = p1;
  }
}

// ---- Shared FFN prologue macro: tile lists + X staging ----------------------
#define FFN_PROLOGUE(IDXB, WGTB, X)                                            \
  const int e    = blockIdx.y;                                                 \
  const int cE   = cnt[e];                                                     \
  const int base = blockIdx.x * MT;                                            \
  if (base >= cE) return;                                                      \
  const int tid  = threadIdx.x;                                                \
  const int wave = tid >> 5;                                                   \
  const int lane = tid & 31;                                                   \
  if (tid < MT) {                                                              \
    int s = base + tid;                                                        \
    bool v = s < cE;                                                           \
    tok_s[tid] = v ? (IDXB)[e * N_ + s] : 0;                                   \
    wgt_s[tid] = v ? (WGTB)[e * N_ + s] : 0.f;                                 \
  }                                                                            \
  __syncthreads();                                                             \
  for (int i = tid; i < MT * D_ / 2; i += 256) {                               \
    int r = i >> 9, c = (i & 511) * 2;                                         \
    float2 xv = *(const float2*)&(X)[(size_t)tok_s[r] * D_ + c];               \
    Xs[r][c] = (__bf16)xv.x; Xs[r][c + 1] = (__bf16)xv.y;                      \
  }                                                                            \
  __syncthreads();                                                             \
  const int hi8 = (lane & 16) ? 8 : 0;                                         \
  float wrow[8];                                                               \
  _Pragma("unroll")                                                            \
  for (int r = 0; r < 8; ++r) wrow[r] = wgt_s[r + hi8];                        \
  v8f oacc[8] = {};                                                            \
  const __bf16* arow = &Xs[lane & 15][0];                                      \
  const __bf16* hrow = &Hs[lane & 15][0];                                      \
  const int cbase  = wave * 128;                                               \
  const int fslice = wave * 32;

#define FFN_EPILOGUE(OUT)                                                      \
  _Pragma("unroll")                                                            \
  for (int t = 0; t < 8; ++t) {                                                \
    _Pragma("unroll")                                                          \
    for (int r = 0; r < 8; ++r) {                                              \
      if (wrow[r] != 0.f) {                                                    \
        float* dst = &(OUT)[(size_t)tok_s[r + hi8] * D_ + cbase + t * 16 + (lane & 15)]; \
        __hip_atomic_fetch_add(dst, oacc[t][r], __ATOMIC_RELAXED,              \
                               __HIP_MEMORY_SCOPE_AGENT);                      \
      }                                                                        \
    }                                                                          \
  }

// ---- Kernel 2a: fused grouped SwiGLU FFN, pre-transposed bf16 weights ------
__global__ __launch_bounds__(256, 1)
void moe_ffn_bf16w(const float* __restrict__ x,
                   const __bf16* __restrict__ w1t,  // [E][F][D]
                   const __bf16* __restrict__ w3t,  // [E][F][D]
                   const __bf16* __restrict__ w2t,  // [E][D][F]
                   const int* __restrict__ cnt, const int* __restrict__ idxb,
                   const float* __restrict__ wgtb, float* __restrict__ out) {
  __shared__ __attribute__((aligned(16))) __bf16 Xs[MT][D_ + 8];
  __shared__ __attribute__((aligned(16))) __bf16 Hs[MT][FCHUNK + 8];
  __shared__ int   tok_s[MT];
  __shared__ float wgt_s[MT];
  FFN_PROLOGUE(idxb, wgtb, x)

  const __bf16* w1e = w1t + (size_t)e * D_ * F_;
  const __bf16* w3e = w3t + (size_t)e * D_ * F_;
  const __bf16* w2e = w2t + (size_t)e * F_ * D_;

  for (int fc = 0; fc < F_; fc += FCHUNK) {
    // phase 1: H[:, fslice..+32) = silu(X@w1) * (X@w3) * wrow
    v8f a1[2] = {}, a3[2] = {};
    for (int k = 0; k < D_; k += 32) {
      v16bf av = ld_a16(arow, k, lane);
#pragma unroll
      for (int t = 0; t < 2; ++t) {
        const int col = fc + fslice + t * 16;            // f index (row of Wt)
        v16bf b1 = ld_b_t(w1e + (size_t)col * D_ + k, D_, lane);
        a1[t] = __builtin_amdgcn_wmma_f32_16x16x32_bf16(false, av, false, b1,
                                                        (short)0, a1[t], false, false);
        v16bf b3 = ld_b_t(w3e + (size_t)col * D_ + k, D_, lane);
        a3[t] = __builtin_amdgcn_wmma_f32_16x16x32_bf16(false, av, false, b3,
                                                        (short)0, a3[t], false, false);
      }
    }
#pragma unroll
    for (int t = 0; t < 2; ++t) {
#pragma unroll
      for (int r = 0; r < 8; ++r) {
        float h = silu_fast(a1[t][r]) * a3[t][r] * wrow[r];
        Hs[r + hi8][fslice + t * 16 + (lane & 15)] = (__bf16)h;
      }
    }
    __syncthreads();
    // phase 2: oacc += H_chunk @ w2[fc..fc+256, cbase..cbase+128)
    for (int ks = 0; ks < FCHUNK; ks += 32) {
      v16bf av = ld_a16(hrow, ks, lane);
#pragma unroll
      for (int t = 0; t < 8; ++t) {
        const int col = cbase + t * 16;                  // d index (row of W2t)
        v16bf b = ld_b_t(w2e + (size_t)col * F_ + fc + ks, F_, lane);
        oacc[t] = __builtin_amdgcn_wmma_f32_16x16x32_bf16(false, av, false, b,
                                                          (short)0, oacc[t], false, false);
      }
    }
    __syncthreads();
  }
  FFN_EPILOGUE(out)
}

// ---- Kernel 2b: fallback, direct fp32 gathers (small workspace) ------------
__global__ __launch_bounds__(256, 1)
void moe_ffn_f32w(const float* __restrict__ x,  const float* __restrict__ w1,
                  const float* __restrict__ w3, const float* __restrict__ w2,
                  const int* __restrict__ cnt,  const int* __restrict__ idxb,
                  const float* __restrict__ wgtb, float* __restrict__ out) {
  __shared__ __attribute__((aligned(16))) __bf16 Xs[MT][D_ + 8];
  __shared__ __attribute__((aligned(16))) __bf16 Hs[MT][FCHUNK + 8];
  __shared__ int   tok_s[MT];
  __shared__ float wgt_s[MT];
  FFN_PROLOGUE(idxb, wgtb, x)

  const float* w1e = w1 + (size_t)e * D_ * F_;
  const float* w3e = w3 + (size_t)e * D_ * F_;
  const float* w2e = w2 + (size_t)e * F_ * D_;

  for (int fc = 0; fc < F_; fc += FCHUNK) {
    v8f a1[2] = {}, a3[2] = {};
    for (int k = 0; k < D_; k += 32) {
      v16bf av = ld_a16(arow, k, lane);
      __builtin_prefetch(w1e + (size_t)(k + 32) * F_ + fc + fslice, 0, 3);
      __builtin_prefetch(w3e + (size_t)(k + 32) * F_ + fc + fslice, 0, 3);
#pragma unroll
      for (int t = 0; t < 2; ++t) {
        const int col = fc + fslice + t * 16;
        v16bf b1 = ld_b_g32(w1e + (size_t)k * F_ + col, F_, lane);
        a1[t] = __builtin_amdgcn_wmma_f32_16x16x32_bf16(false, av, false, b1,
                                                        (short)0, a1[t], false, false);
        v16bf b3 = ld_b_g32(w3e + (size_t)k * F_ + col, F_, lane);
        a3[t] = __builtin_amdgcn_wmma_f32_16x16x32_bf16(false, av, false, b3,
                                                        (short)0, a3[t], false, false);
      }
    }
#pragma unroll
    for (int t = 0; t < 2; ++t) {
#pragma unroll
      for (int r = 0; r < 8; ++r) {
        float h = silu_fast(a1[t][r]) * a3[t][r] * wrow[r];
        Hs[r + hi8][fslice + t * 16 + (lane & 15)] = (__bf16)h;
      }
    }
    __syncthreads();
    for (int ks = 0; ks < FCHUNK; ks += 32) {
      v16bf av = ld_a16(hrow, ks, lane);
#pragma unroll
      for (int t = 0; t < 8; ++t) {
        v16bf b = ld_b_g32(w2e + (size_t)(fc + ks) * D_ + cbase + t * 16, D_, lane);
        oacc[t] = __builtin_amdgcn_wmma_f32_16x16x32_bf16(false, av, false, b,
                                                          (short)0, oacc[t], false, false);
      }
    }
    __syncthreads();
  }
  FFN_EPILOGUE(out)
}

// ---- Host launcher ----------------------------------------------------------
extern "C" void kernel_launch(void* const* d_in, const int* in_sizes, int n_in,
                              void* d_out, int out_size, void* d_ws, size_t ws_size,
                              hipStream_t stream) {
  (void)in_sizes; (void)n_in; (void)out_size;
  const float* x  = (const float*)d_in[0];   // [N, D]
  const float* gw = (const float*)d_in[1];   // [D, E]
  const float* w1 = (const float*)d_in[2];   // [E, D, F]
  const float* w3 = (const float*)d_in[3];   // [E, D, F]
  const float* w2 = (const float*)d_in[4];   // [E, F, D]

  float* out_y  = (float*)d_out;             // [N, D]
  float* logits = out_y + (size_t)N_ * D_;   // [N, E] (tuple concat)

  // Workspace layout: counters + per-expert lists (~512 KB), then (optionally)
  // pre-transposed bf16 weights (3 x 32 MB).
  char*  wsb  = (char*)d_ws;
  int*   cnt  = (int*)wsb;                   // [16]
  int*   idxb = cnt + 16;                    // [E, N]
  float* wgtb = (float*)(idxb + E_ * N_);    // [E, N]
  size_t list_bytes = 16 * 4 + (size_t)E_ * N_ * 4 * 2;
  size_t woff   = (list_bytes + 255) & ~(size_t)255;
  size_t wbytes = (size_t)E_ * D_ * F_ * sizeof(__bf16);   // 32 MB each
  const bool big_ws = ws_size >= woff + 3 * wbytes;        // constant per run

  hipMemsetAsync(d_out, 0, (size_t)N_ * D_ * sizeof(float), stream);
  moe_init<<<1, 32, 0, stream>>>(cnt);
  moe_router<<<N_ / 8, 256, 0, stream>>>(x, gw, logits, cnt, idxb, wgtb);

  if (big_ws) {
    __bf16* w1t = (__bf16*)(wsb + woff);
    __bf16* w3t = w1t + (size_t)E_ * D_ * F_;
    __bf16* w2t = w3t + (size_t)E_ * D_ * F_;
    transpose_cvt<<<dim3(F_ / 32, D_ / 32, E_), 256, 0, stream>>>(w1, w1t, D_, F_);
    transpose_cvt<<<dim3(F_ / 32, D_ / 32, E_), 256, 0, stream>>>(w3, w3t, D_, F_);
    transpose_cvt<<<dim3(D_ / 32, F_ / 32, E_), 256, 0, stream>>>(w2, w2t, F_, D_);
    moe_ffn_bf16w<<<dim3(N_ / MT, E_), 256, 0, stream>>>(x, w1t, w3t, w2t,
                                                         cnt, idxb, wgtb, out_y);
  } else {
    moe_ffn_f32w<<<dim3(N_ / MT, E_), 256, 0, stream>>>(x, w1, w3, w2,
                                                        cnt, idxb, wgtb, out_y);
  }
}
